// FastRCNN_47699906789451
// MI455X (gfx1250) — matrix operand order repaired and checked by reading, not compiled, not run on hardware
//
#include <hip/hip_runtime.h>
#include <hip/hip_bf16.h>

typedef _Float16 v16h __attribute__((ext_vector_type(16)));
typedef _Float16 v8h  __attribute__((ext_vector_type(8)));
typedef float    v8f  __attribute__((ext_vector_type(8)));

#define BN_EPS 1e-5f

// ---------------------------------------------------------------------------
// Fused conv3x3(SAME) + bias + BN(inference) + ReLU, implicit GEMM via WMMA.
// Block = 128 threads (4 waves). Wave w computes a 16(Cout) x 16(x-pixels)
// output tile on row y0+w. K = Cin*9 consumed 32 at a time; per chunk the
// block stages a (6 rows x 18 cols x 32 cin) f16 input tile and a
// (16 cout x 9 tap x 32 cin) f16 weight slab in LDS, then does 9 WMMAs.
// ---------------------------------------------------------------------------
__global__ __launch_bounds__(128)
void conv3x3_bn_relu_wmma(const float* __restrict__ in, const float* __restrict__ wgt,
                          const float* __restrict__ bias, const float* __restrict__ gamma,
                          const float* __restrict__ beta, const float* __restrict__ mean,
                          const float* __restrict__ var,
                          float* __restrict__ out, int Cin, int Cout, int H, int W)
{
    __shared__ _Float16 lds_in[6 * 18 * 32];   // [row][col][cin]  (cin innermost)
    __shared__ _Float16 lds_w [16 * 9 * 32];   // [co_l][tap][cin]

    const int tid  = threadIdx.x;
    const int wave = tid >> 5;
    const int lane = tid & 31;
    const int n    = lane & 15;     // output column (pixel) / weight row
    const int half = lane >> 4;     // which half-wave (K split)

    const int x0     = blockIdx.x * 16;
    const int y0     = blockIdx.y * 4;
    const int ctiles = Cout >> 4;
    const int bi     = blockIdx.z / ctiles;
    const int co0    = (blockIdx.z % ctiles) << 4;

    const long inBase = (long)bi * Cin * H * W;

    v8f acc = {0.f, 0.f, 0.f, 0.f, 0.f, 0.f, 0.f, 0.f};

    for (int cb = 0; cb < Cin; cb += 32) {
        const int cn = (Cin - cb) < 32 ? (Cin - cb) : 32;
        __syncthreads();

        // ---- stage input halo tile: rows y0-1..y0+4, cols x0-1..x0+16 ----
        for (int idx = tid; idx < 108 * 32; idx += 128) {
            int c    = idx / 108;          // cin within chunk
            int slot = idx - c * 108;      // consecutive threads -> consecutive cols (coalesced)
            int row  = slot / 18;
            int col  = slot - row * 18;
            int gy   = y0 - 1 + row;
            int gx   = x0 - 1 + col;
            float v = 0.f;
            if (c < cn && gy >= 0 && gy < H && gx >= 0 && gx < W)
                v = in[inBase + (long)(cb + c) * H * W + (long)gy * W + gx];
            lds_in[slot * 32 + c] = (_Float16)v;
        }
        // ---- stage weights: w[co][cin][ky][kx], K ordered (tap, cin) in LDS ----
        for (int idx = tid; idx < 16 * 9 * 32; idx += 128) {
            int co_l = idx / 288;
            int rem  = idx - co_l * 288;
            int r    = rem >> 5;           // tap = ky*3+kx
            int c    = rem & 31;
            float v = 0.f;
            if (c < cn)
                v = wgt[(long)(co0 + co_l) * Cin * 9 + (long)(cb + c) * 9 + r];
            lds_w[idx] = (_Float16)v;
        }
        __syncthreads();

        // ---- 9 WMMAs: one per 3x3 tap, K = 32 cins each ----
        for (int r = 0; r < 9; ++r) {
            int ky = r / 3, kx = r - ky * 3;
            int rr = wave + ky;            // LDS row (row 0 == y0-1)
            int cc = n + kx;               // LDS col (col 0 == x0-1)

            // B fragment (32x16): lane half h holds K = 16h+0..15 (contiguous cins)
            const _Float16* bp = &lds_in[(rr * 18 + cc) * 32 + half * 16];
            v8h blo = *(const v8h*)(bp);
            v8h bhi = *(const v8h*)(bp + 8);
            v16h bf = __builtin_shufflevector(blo, bhi,
                0,1,2,3,4,5,6,7,8,9,10,11,12,13,14,15);

            // A fragment (16x32): lane half h holds K = 16h+0..15 (contiguous cins)
            const _Float16* ap = &lds_w[(n * 9 + r) * 32];
            v8h alo = *(const v8h*)(ap + half * 16);
            v8h ahi = *(const v8h*)(ap + half * 16 + 8);
            v16h af = __builtin_shufflevector(alo, ahi,
                0,1,2,3,4,5,6,7,8,9,10,11,12,13,14,15);

            acc = __builtin_amdgcn_wmma_f32_16x16x32_f16(
                false, af, false, bf, (short)0, acc, false, false);
        }
    }

    // ---- fused bias + BN + ReLU epilogue.  D layout: VGPR j, M = j + 8*half ----
    const int y = y0 + wave;
    const int x = x0 + n;
    #pragma unroll
    for (int j = 0; j < 8; ++j) {
        int cg = co0 + j + half * 8;
        float sc = gamma[cg] * rsqrtf(var[cg] + BN_EPS);
        float sh = (bias[cg] - mean[cg]) * sc + beta[cg];
        float v  = acc[j] * sc + sh;
        v = fmaxf(v, 0.f);
        out[((long)bi * Cout + cg) * H * W + (long)y * W + x] = v;
    }
}

// ---------------------------------------------------------------------------
// 2x2 stride-2 max pool. One thread per output element; (b,c) fused.
// ---------------------------------------------------------------------------
__global__ void maxpool2x2(const float* __restrict__ in, float* __restrict__ out,
                           int H, int W, int total)
{
    int idx = blockIdx.x * blockDim.x + threadIdx.x;
    if (idx >= total) return;
    int Ho = H >> 1, Wo = W >> 1;
    int x = idx % Wo;
    int y = (idx / Wo) % Ho;
    int c = idx / (Wo * Ho);               // fused b*C + c
    const float* p = in + ((long)c * H + 2 * y) * W + 2 * x;
    out[idx] = fmaxf(fmaxf(p[0], p[1]), fmaxf(p[W], p[W + 1]));
}

// ---------------------------------------------------------------------------
// ROI adaptive max pool 4x4 over [2,512,16,16] features.
// torch bins: [floor(i*L/4), ceil((i+1)*L/4)).  Output flattened (C,H,W).
// ---------------------------------------------------------------------------
__global__ void roi_adamaxpool(const float* __restrict__ feat, const int* __restrict__ boxes,
                               float* __restrict__ pooled)
{
    int idx = blockIdx.x * blockDim.x + threadIdx.x;
    if (idx >= 128 * 512 * 16) return;
    int bin  = idx & 15;
    int c    = (idx >> 4) & 511;
    int nroi = idx >> 13;
    int ix = bin & 3, iy = bin >> 2;

    const int* bx = boxes + nroi * 5;
    int img = bx[0], lx = bx[1], ly = bx[2], rx = bx[3], ry = bx[4];
    int Lx = rx - lx, Ly = ry - ly;
    int bsx = lx + (ix * Lx) / 4;
    int bex = lx + ((ix + 1) * Lx + 3) / 4;
    int bsy = ly + (iy * Ly) / 4;
    int bey = ly + ((iy + 1) * Ly + 3) / 4;

    const float* f = feat + ((long)img * 512 + c) * 256;   // 16x16 plane
    float m = -3.402823466e38f;
    for (int y = bsy; y < bey; ++y)
        for (int x = bsx; x < bex; ++x)
            m = fmaxf(m, f[y * 16 + x]);
    pooled[(long)nroi * 8192 + c * 16 + iy * 4 + ix] = m;
}

// ---------------------------------------------------------------------------
// Dense layer: out[n,o] = in[n,:] . w[o,:] + b[o] (optional ReLU).
// Tiny FLOP count vs convs -> simple VALU kernel is fine.
// ---------------------------------------------------------------------------
__global__ void fc_kernel(const float* __restrict__ in, const float* __restrict__ w,
                          const float* __restrict__ b, float* __restrict__ out,
                          int N, int K, int O, int relu)
{
    int idx = blockIdx.x * blockDim.x + threadIdx.x;
    if (idx >= N * O) return;
    int o = idx % O, n = idx / O;
    const float* xi = in + (long)n * K;
    const float* wi = w  + (long)o * K;
    float s = b[o];
    for (int k = 0; k < K; ++k) s += xi[k] * wi[k];
    if (relu) s = fmaxf(s, 0.f);
    out[idx] = s;
}

// ---------------------------------------------------------------------------
// Host-side orchestration
// ---------------------------------------------------------------------------
extern "C" void kernel_launch(void* const* d_in, const int* in_sizes, int n_in,
                              void* d_out, int out_size, void* d_ws, size_t ws_size,
                              hipStream_t stream)
{
    (void)in_sizes; (void)n_in; (void)out_size; (void)ws_size;

    const float* images = (const float*)d_in[0];
    const int*   boxes  = (const int*)d_in[1];
    float* outp = (float*)d_out;

    // ws layout (floats): two ping-pong conv buffers sized for the largest
    // activation (2*64*256*256), FC scratch carved out of bufB after features
    // have migrated to bufA.
    float* bufA = (float*)d_ws;
    float* bufB = bufA + 8388608;

    const int B = 2;

    auto conv = [&](int li, const float* in, float* out,
                    int Cin, int Cout, int H, int W) {
        int base = 2 + 6 * li;
        dim3 grid(W / 16, H / 4, B * (Cout / 16));
        conv3x3_bn_relu_wmma<<<grid, 128, 0, stream>>>(
            in,
            (const float*)d_in[base + 0],   // w
            (const float*)d_in[base + 1],   // b
            (const float*)d_in[base + 2],   // g
            (const float*)d_in[base + 3],   // be
            (const float*)d_in[base + 4],   // m
            (const float*)d_in[base + 5],   // v
            out, Cin, Cout, H, W);
    };

    auto pool = [&](const float* in, float* out, int C, int H, int W) {
        int total = B * C * (H / 2) * (W / 2);
        maxpool2x2<<<(total + 255) / 256, 256, 0, stream>>>(in, out, H, W, total);
    };

    // ---- VGG16-bn features (final maxpool removed) ----
    conv(0, images, bufA,   3,  64, 256, 256);
    conv(1, bufA,   bufB,  64,  64, 256, 256);
    pool(bufB, bufA, 64, 256, 256);                  // -> 128
    conv(2, bufA,   bufB,  64, 128, 128, 128);
    conv(3, bufB,   bufA, 128, 128, 128, 128);
    pool(bufA, bufB, 128, 128, 128);                 // -> 64
    conv(4, bufB,   bufA, 128, 256,  64,  64);
    conv(5, bufA,   bufB, 256, 256,  64,  64);
    conv(6, bufB,   bufA, 256, 256,  64,  64);
    pool(bufA, bufB, 256, 64, 64);                   // -> 32
    conv(7, bufB,   bufA, 256, 512,  32,  32);
    conv(8, bufA,   bufB, 512, 512,  32,  32);
    conv(9, bufB,   bufA, 512, 512,  32,  32);
    pool(bufA, bufB, 512, 32, 32);                   // -> 16
    conv(10, bufB,  bufA, 512, 512,  16,  16);
    conv(11, bufA,  bufB, 512, 512,  16,  16);
    conv(12, bufB,  bufA, 512, 512,  16,  16);
    // features: bufA = [2,512,16,16]

    // ---- ROI adaptive max pool -> [128, 8192] (C,H,W flatten order) ----
    float* pooled = bufB;                       // 128*8192 = 1,048,576
    float* t1 = bufB + 1048576;                 // fc1 out : 128*16
    float* t2 = t1 + 2048;                      // fc2 out : 128*4096
    float* t3 = t2 + 524288;                    // fc3 out : 128*32
    float* t4 = t3 + 4096;                      // fc4 out : 128*4096
    {
        int total = 128 * 512 * 16;
        roi_adamaxpool<<<(total + 255) / 256, 256, 0, stream>>>(bufA, boxes, pooled);
    }

    // ---- FC head ----
    const float* fc1w = (const float*)d_in[80]; const float* fc1b = (const float*)d_in[81];
    const float* fc2w = (const float*)d_in[82]; const float* fc2b = (const float*)d_in[83];
    const float* fc3w = (const float*)d_in[84]; const float* fc3b = (const float*)d_in[85];
    const float* fc4w = (const float*)d_in[86]; const float* fc4b = (const float*)d_in[87];
    const float* clsw = (const float*)d_in[88]; const float* clsb = (const float*)d_in[89];
    const float* regw = (const float*)d_in[90]; const float* regb = (const float*)d_in[91];

    auto fc = [&](const float* in, const float* w, const float* b, float* out,
                  int N, int K, int O, int relu) {
        int total = N * O;
        fc_kernel<<<(total + 255) / 256, 256, 0, stream>>>(in, w, b, out, N, K, O, relu);
    };

    fc(pooled, fc1w, fc1b, t1, 128, 8192,   16, 0);
    fc(t1,     fc2w, fc2b, t2, 128,   16, 4096, 1);
    fc(t2,     fc3w, fc3b, t3, 128, 4096,   32, 0);
    fc(t3,     fc4w, fc4b, t4, 128,   32, 4096, 1);
    fc(t4,     clsw, clsb, outp,        128, 4096, 21, 0);   // cls -> d_out[0..2688)
    fc(t4,     regw, regb, outp + 2688, 128, 4096,  4, 0);   // reg -> d_out[2688..3200)
}
